// GCN_30391188586774
// MI455X (gfx1250) — matrix operand order repaired and checked by reading, not compiled, not run on hardware
//
#include <hip/hip_runtime.h>
#include <hip/hip_bf16.h>
#include <stdint.h>

// Problem constants (from reference)
#define N_NODES 100000
#define N_EDGES 3200000
#define IN_F    256
#define HID     256
#define CLS     64
#define KDIM    256          // GEMM K is 256 for all three layers
#define LN_EPS  1e-5f

// CDNA5 WMMA vector types (wave32)
typedef __attribute__((ext_vector_type(16))) __bf16 v16bf;
typedef __attribute__((ext_vector_type(8)))  __bf16 v8bf;
typedef __attribute__((ext_vector_type(8)))  float  v8f;

__device__ __forceinline__ unsigned short f2bf_bits(float x) {
    // round-to-nearest-even float -> bf16 bits
    unsigned u = __float_as_uint(x);
    u += 0x7FFFu + ((u >> 16) & 1u);
    return (unsigned short)(u >> 16);
}

__device__ __forceinline__ v8f wmma_bf16(v16bf a, v16bf b, v8f c) {
    return __builtin_amdgcn_wmma_f32_16x16x32_bf16(
        /*neg_a=*/false, a, /*neg_b=*/false, b,
        /*c_mod=*/(short)0, c, /*reuse_a=*/false, /*reuse_b=*/false);
}

// ---------------- utility kernels ----------------

__global__ void gcn_zero_f32(float* __restrict__ p, long n) {
    long stride = (long)gridDim.x * blockDim.x;
    for (long i = (long)blockIdx.x * blockDim.x + threadIdx.x; i < n; i += stride)
        p[i] = 0.0f;
}

__global__ void gcn_degrees(const int* __restrict__ src, const int* __restrict__ dst,
                            float* __restrict__ degS, float* __restrict__ degD, int e_cnt) {
    int i = blockIdx.x * blockDim.x + threadIdx.x;
    if (i < e_cnt) {
        atomicAdd(&degS[src[i]], 1.0f);
        atomicAdd(&degD[dst[i]], 1.0f);
    }
}

__global__ void gcn_norm_inplace(float* __restrict__ deg, int n) {
    int i = blockIdx.x * blockDim.x + threadIdx.x;
    if (i < n) deg[i] = rsqrtf(fmaxf(deg[i], 1.0f));
}

// ---------------- edge scatter-add (SpMM), one wave per edge ----------------
// agg[dst] += X[src] * norm_src[src]; 256 f32 features per edge, lane covers 8.
__global__ __launch_bounds__(256) void gcn_edge_agg(
    const float* __restrict__ X, const int* __restrict__ src, const int* __restrict__ dst,
    const float* __restrict__ normS, float* __restrict__ agg, int e_cnt)
{
    int e = (int)(((long)blockIdx.x * blockDim.x + threadIdx.x) >> 5);  // wave32
    if (e >= e_cnt) return;                                             // wave-uniform
    int lane = threadIdx.x & 31;
    int s = src[e], d = dst[e];
    float ns = normS[s];
    const float4* xs = reinterpret_cast<const float4*>(X + (size_t)s * HID) + lane;
    float4 v0 = xs[0];        // cols [lane*4, lane*4+4)
    float4 v1 = xs[32];       // cols [128+lane*4, ...)
    float* ad = agg + (size_t)d * HID + lane * 4;
    atomicAdd(ad + 0,   v0.x * ns);
    atomicAdd(ad + 1,   v0.y * ns);
    atomicAdd(ad + 2,   v0.z * ns);
    atomicAdd(ad + 3,   v0.w * ns);
    atomicAdd(ad + 128, v1.x * ns);
    atomicAdd(ad + 129, v1.y * ns);
    atomicAdd(ad + 130, v1.z * ns);
    atomicAdd(ad + 131, v1.w * ns);
}

// ---------------- scale by norm_dst and convert to bf16 ----------------
__global__ void gcn_scale_cvt(const float* __restrict__ agg, const float* __restrict__ normD,
                              unsigned short* __restrict__ abf, long n) {
    long stride = (long)gridDim.x * blockDim.x;
    for (long i = (long)blockIdx.x * blockDim.x + threadIdx.x; i < n; i += stride) {
        int row = (int)(i >> 8);  // HID == 256
        abf[i] = f2bf_bits(agg[i] * normD[row]);
    }
}

// ---------------- transpose-convert W (K x Nn, row-major f32) -> Wt (Nn x K, bf16) ----------------
__global__ void gcn_cvt_wT(const float* __restrict__ W, unsigned short* __restrict__ Wt,
                           int K, int Nn) {
    int idx = blockIdx.x * blockDim.x + threadIdx.x;
    if (idx >= K * Nn) return;
    int n = idx / K, k = idx % K;
    Wt[idx] = f2bf_bits(W[k * Nn + n]);
}

// ---------------- bf16 WMMA GEMM: C[M,Nn] = A[M,K] * W[K,Nn] + bias (opt ReLU) ----------------
// One wave computes a 32x32 output block: 2 A-fragments x 2 B-fragments -> 4 accumulators.
// K is hardcoded (256) so the k-loop fully unrolls: 64 b128 loads + 32 WMMAs, no branches.
__global__ __launch_bounds__(256) void gcn_gemm_wmma(
    const unsigned short* __restrict__ A,   // bf16 bits, row-major [M,KDIM]
    const unsigned short* __restrict__ Wt,  // bf16 bits, row-major [Nn,KDIM] (W transposed)
    const float* __restrict__ bias,
    float* __restrict__ C,                  // f32 row-major [M,ldc]
    int Mg, int Ng, int ldc, int relu)      // Mg = M/32 groups, Ng = Nn/32 groups
{
    int gwave = (int)(((long)blockIdx.x * blockDim.x + threadIdx.x) >> 5);
    if (gwave >= Mg * Ng) return;           // wave-uniform: EXEC stays all-1s for WMMA
    int lane = threadIdx.x & 31;
    int mg = gwave / Ng, ng = gwave % Ng;
    int m0 = mg << 5, n0 = ng << 5;
    int lrow = lane & 15;

    // A fragment: lane = row; K = 16 contiguous starting at 0 (lanes<16) / 16
    const unsigned short* ap0 = A  + (size_t)(m0 + lrow) * KDIM + ((lane < 16) ? 0 : 16);
    const unsigned short* ap1 = ap0 + 16 * KDIM;
    // B fragment: lane = col; K = 16 contiguous starting at 0 (lanes<16) / 16
    const unsigned short* bp0 = Wt + (size_t)(n0 + lrow) * KDIM + ((lane < 16) ? 0 : 16);
    const unsigned short* bp1 = bp0 + 16 * KDIM;

    v8f acc00 = {}, acc01 = {}, acc10 = {}, acc11 = {};
    #pragma unroll
    for (int kt = 0; kt < KDIM / 32; ++kt) {
        union { v16bf v; v8bf h[2]; } a0, a1;
        a0.h[0] = *reinterpret_cast<const v8bf*>(ap0 + kt * 32);
        a0.h[1] = *reinterpret_cast<const v8bf*>(ap0 + kt * 32 + 8);
        a1.h[0] = *reinterpret_cast<const v8bf*>(ap1 + kt * 32);
        a1.h[1] = *reinterpret_cast<const v8bf*>(ap1 + kt * 32 + 8);
        v16bf b0 = *reinterpret_cast<const v16bf*>(bp0 + kt * 32);
        v16bf b1 = *reinterpret_cast<const v16bf*>(bp1 + kt * 32);
        acc00 = wmma_bf16(a0.v, b0, acc00);
        acc01 = wmma_bf16(a0.v, b1, acc01);
        acc10 = wmma_bf16(a1.v, b0, acc10);
        acc11 = wmma_bf16(a1.v, b1, acc11);
    }

    // C/D layout: VGPR r -> row r + 8*(lane>=16), col = lane&15
    float bv0 = bias[n0 + lrow];
    float bv1 = bias[n0 + 16 + lrow];
    int mrow = m0 + ((lane < 16) ? 0 : 8);
    #pragma unroll
    for (int r = 0; r < 8; ++r) {
        float v00 = acc00[r] + bv0;
        float v01 = acc01[r] + bv1;
        float v10 = acc10[r] + bv0;
        float v11 = acc11[r] + bv1;
        if (relu) {
            v00 = fmaxf(v00, 0.0f); v01 = fmaxf(v01, 0.0f);
            v10 = fmaxf(v10, 0.0f); v11 = fmaxf(v11, 0.0f);
        }
        float* c0 = C + (size_t)(mrow + r) * ldc + n0 + lrow;
        float* c1 = C + (size_t)(mrow + 16 + r) * ldc + n0 + lrow;
        c0[0]  = v00;
        c0[16] = v01;
        c1[0]  = v10;
        c1[16] = v11;
    }
}

// ---------------- full-tensor layernorm (two kernels + 2-scalar reduction) ----------------
__global__ __launch_bounds__(256) void gcn_ln_reduce(const float* __restrict__ h, long n,
                                                     float* __restrict__ scal) {
    __shared__ float ss[256];
    __shared__ float sq[256];
    float s = 0.0f, q = 0.0f;
    long stride = (long)gridDim.x * blockDim.x;
    for (long i = (long)blockIdx.x * blockDim.x + threadIdx.x; i < n; i += stride) {
        float v = h[i];
        s += v;
        q = fmaf(v, v, q);
    }
    int t = threadIdx.x;
    ss[t] = s; sq[t] = q;
    __syncthreads();
    for (int w = 128; w > 0; w >>= 1) {
        if (t < w) { ss[t] += ss[t + w]; sq[t] += sq[t + w]; }
        __syncthreads();
    }
    if (t == 0) {
        atomicAdd(&scal[0], ss[0]);
        atomicAdd(&scal[1], sq[0]);
    }
}

__global__ void gcn_ln_norm(float* __restrict__ h, long n, const float* __restrict__ scal) {
    float inv_n = 1.0f / (float)n;
    float mu  = scal[0] * inv_n;
    float var = scal[1] * inv_n - mu * mu;
    float r   = rsqrtf(var + LN_EPS);
    long stride = (long)gridDim.x * blockDim.x;
    for (long i = (long)blockIdx.x * blockDim.x + threadIdx.x; i < n; i += stride)
        h[i] = (h[i] - mu) * r;
}

// ---------------- host orchestration ----------------
extern "C" void kernel_launch(void* const* d_in, const int* in_sizes, int n_in,
                              void* d_out, int out_size, void* d_ws, size_t ws_size,
                              hipStream_t stream) {
    (void)in_sizes; (void)n_in; (void)out_size; (void)ws_size;

    const float* features = (const float*)d_in[0];
    const int*   src      = (const int*)d_in[1];
    const int*   dst      = (const int*)d_in[2];
    const float* W0 = (const float*)d_in[3];
    const float* b0 = (const float*)d_in[4];
    const float* W1 = (const float*)d_in[5];
    const float* b1 = (const float*)d_in[6];
    const float* W2 = (const float*)d_in[7];
    const float* b2 = (const float*)d_in[8];
    float* out = (float*)d_out;

    // workspace carve-out (256B aligned)
    char* ws = (char*)d_ws;
    size_t off = 0;
    auto carve = [&](size_t bytes) {
        char* p = ws + off;
        off = (off + bytes + 255) & ~(size_t)255;
        return p;
    };
    float* normS = (float*)carve((size_t)N_NODES * 4);             // deg_out -> norm_src
    float* normD = (float*)carve((size_t)N_NODES * 4);             // deg_in  -> norm_dst
    float* h     = (float*)carve((size_t)N_NODES * HID * 4);       // activations (X/Y)
    float* agg   = (float*)carve((size_t)N_NODES * HID * 4);       // scatter-add target
    unsigned short* abf = (unsigned short*)carve((size_t)N_NODES * HID * 2);  // bf16 A
    unsigned short* wt  = (unsigned short*)carve((size_t)IN_F * HID * 2);     // bf16 W^T
    float* scal  = (float*)carve(256);                             // LN sum / sumsq

    const int  BLK = 256;
    const long NH  = (long)N_NODES * HID;

    // degree norms (D^-1/2)
    gcn_zero_f32<<<1024, BLK, 0, stream>>>(normS, N_NODES);
    gcn_zero_f32<<<1024, BLK, 0, stream>>>(normD, N_NODES);
    gcn_degrees<<<(N_EDGES + BLK - 1) / BLK, BLK, 0, stream>>>(src, dst, normS, normD, N_EDGES);
    gcn_norm_inplace<<<(N_NODES + BLK - 1) / BLK, BLK, 0, stream>>>(normS, N_NODES);
    gcn_norm_inplace<<<(N_NODES + BLK - 1) / BLK, BLK, 0, stream>>>(normD, N_NODES);

    auto layer = [&](const float* X, const float* W, const float* bias, float* Y,
                     int ncols, int relu, int do_ln) {
        gcn_zero_f32<<<4096, BLK, 0, stream>>>(agg, NH);
        gcn_edge_agg<<<(N_EDGES + 7) / 8, BLK, 0, stream>>>(X, src, dst, normS, agg, N_EDGES);
        gcn_scale_cvt<<<4096, BLK, 0, stream>>>(agg, normD, abf, NH);
        gcn_cvt_wT<<<(IN_F * ncols + BLK - 1) / BLK, BLK, 0, stream>>>(W, wt, IN_F, ncols);
        int Mg = N_NODES / 32, Ng = ncols / 32;   // 3125 x {8,2}, both exact
        int waves = Mg * Ng;                       // 8 waves per 256-thread block
        gcn_gemm_wmma<<<(waves + 7) / 8, BLK, 0, stream>>>(abf, wt, bias, Y, Mg, Ng, ncols, relu);
        if (do_ln) {
            gcn_zero_f32<<<1, 64, 0, stream>>>(scal, 2);
            gcn_ln_reduce<<<2048, BLK, 0, stream>>>(Y, NH, scal);
            gcn_ln_norm<<<4096, BLK, 0, stream>>>(Y, NH, scal);
        }
    };

    layer(features, W0, b0, h,   HID, /*relu=*/1, /*ln=*/1);
    layer(h,        W1, b1, h,   HID, /*relu=*/1, /*ln=*/1);
    layer(h,        W2, b2, out, CLS, /*relu=*/0, /*ln=*/0);
}
